// BahdanauAttention_57844619542902
// MI455X (gfx1250) — compile-verified, matrix operand order checked by reading, not run on hardware
//
#include <hip/hip_runtime.h>
#include <hip/hip_bf16.h>
#include <math.h>

#define B_ 32
#define T_ 2048
#define D_ 1024
#define U_ 1024

typedef __attribute__((ext_vector_type(16))) __bf16 v16bf;
typedef __attribute__((ext_vector_type(8)))  float  v8f;

union FragBF16 { v16bf v; uint4 q[2]; };
union Pack8    { uint4 q; __bf16 h[8]; };
union Pack4    { uint2 d; __bf16 h[4]; };

// ---------------------------------------------------------------- utilities
__global__ __launch_bounds__(256) void zero_kernel(float* score, float* out) {
    int i = blockIdx.x * 256 + threadIdx.x;      // 96*256 = 24576 threads
    #pragma unroll
    for (int j = 0; j < 4; ++j) {
        int idx = i * 4 + j;                     // 0 .. 98303
        if (idx < B_ * T_)      score[idx] = 0.0f;
        else                    out[idx - B_ * T_] = 0.0f;
    }
}

// Pre-pack W2 (fp32, [D][U]) into bf16 *B-fragment order*:
//   packed[(ut*32 + kstep)*32 + lane][16 bf16]
// per ISA 16-bit B layout: lane&15 -> column, lane>>4 selects K 0-7/16-23 vs 8-15/24-31.
__global__ __launch_bounds__(256) void pack_w2_kernel(const float* __restrict__ w2,
                                                      __bf16* __restrict__ dst) {
    int idx   = blockIdx.x * 256 + threadIdx.x;  // 65536 threads
    int lane  = idx & 31;
    int kstep = (idx >> 5) & 31;
    int ut    = idx >> 10;                       // 0..63 (16-col tile)
    int n     = ut * 16 + (lane & 15);
    int kbase = kstep * 32 + ((lane >> 4) << 3);
    Pack8 lo, hi;
    #pragma unroll
    for (int e = 0; e < 8; ++e) {
        lo.h[e] = (__bf16)w2[(size_t)(kbase + e)      * U_ + n];
        hi.h[e] = (__bf16)w2[(size_t)(kbase + 16 + e) * U_ + n];
    }
    uint4* d = (uint4*)(dst + (size_t)idx * 16);
    d[0] = lo.q;
    d[1] = hi.q;
}

__global__ __launch_bounds__(256) void qproj_kernel(const float* __restrict__ query,
                                                    const float* __restrict__ W1,
                                                    const float* __restrict__ W1b,
                                                    float* __restrict__ qproj) {
    int b = blockIdx.x >> 2;
    int u = ((blockIdx.x & 3) << 8) + threadIdx.x;
    float acc = W1b[u];
    const float* q = query + (size_t)b * D_;
    const float* w = W1 + u;
    for (int d = 0; d < D_; ++d) acc = fmaf(q[d], w[(size_t)d * U_], acc);
    qproj[(size_t)b * U_ + u] = acc;
}

// ------------------------------------------------- main bf16 WMMA score GEMM
// Block tile: M=64 (t rows) x N=128 (u cols), K=1024 in 32 steps of 32.
// 8 wave32s in a 2x4 grid; each wave computes a 32x32 sub-tile (4 accumulators,
// 4 WMMAs per K-step). A staged via double-buffered LDS (1 barrier/step);
// B fragments loaded directly from L2-resident pre-packed W2.
__global__ __launch_bounds__(256)
void bahdanau_score_wmma(const float* __restrict__ values,
                         const __bf16* __restrict__ w2pack,
                         const float* __restrict__ qproj,
                         const float* __restrict__ w2b,
                         const float* __restrict__ vk,
                         float* __restrict__ score) {
    __shared__ __bf16 Abf[2][64][40];   // double-buffered, rows padded to 80B

    const int b     = blockIdx.z;
    const int trow0 = blockIdx.y * 64;
    const int ucol0 = blockIdx.x * 128;
    const int tid   = threadIdx.x;
    const int lane  = tid & 31;
    const int wave  = tid >> 5;
    const int mw    = (wave & 1) * 32;   // wave M offset within block tile
    const int nw    = (wave >> 1) * 32;  // wave N offset within block tile

    v8f c[2][2] = {};

    // A staging: 64 rows x 32 K fp32 per step = 512 float4; 2 per thread
    const int r0 = tid >> 3;             // 0..31
    const int r1 = r0 + 32;              // 32..63
    const int kk = (tid & 7) << 2;       // 0,4,..,28
    const size_t vrow0 = ((size_t)b * T_ + trow0 + r0) * D_ + kk;
    const size_t vrow1 = ((size_t)b * T_ + trow0 + r1) * D_ + kk;

    // fragment addressing (ISA 16-bit operand layout, wave32)
    const int lrow = lane & 15;
    const int k0   = (lane >> 4) << 3;

    // B fragment tile index base for this wave (two 16-col tiles)
    const int ut0 = (ucol0 >> 4) + (wave >> 1) * 2;

    // preload K-step 0
    float4 a0 = *(const float4*)(values + vrow0);
    float4 a1 = *(const float4*)(values + vrow1);

    for (int it = 0; it < 32; ++it) {
        const int p = it & 1;

        // stage current regs into LDS buffer p (fp32 -> bf16)
        Pack4 p0, p1;
        p0.h[0] = (__bf16)a0.x; p0.h[1] = (__bf16)a0.y;
        p0.h[2] = (__bf16)a0.z; p0.h[3] = (__bf16)a0.w;
        p1.h[0] = (__bf16)a1.x; p1.h[1] = (__bf16)a1.y;
        p1.h[2] = (__bf16)a1.z; p1.h[3] = (__bf16)a1.w;
        *(uint2*)&Abf[p][r0][kk] = p0.d;
        *(uint2*)&Abf[p][r1][kk] = p1.d;

        __syncthreads();   // buffer p complete; also guards reuse of p from it-2

        // issue next step's global A loads (latency hidden behind WMMAs)
        if (it + 1 < 32) {
            a0 = *(const float4*)(values + vrow0 + (size_t)(it + 1) * 32);
            a1 = *(const float4*)(values + vrow1 + (size_t)(it + 1) * 32);
            __builtin_prefetch(values + vrow0 + (size_t)(it + 2) * 32, 0, 0);
        }

        // A fragments from LDS (2 per wave, reused across N)
        FragBF16 fa[2], fb[2];
        #pragma unroll
        for (int mi = 0; mi < 2; ++mi) {
            const __bf16* ap = &Abf[p][mw + mi * 16 + lrow][k0];
            fa[mi].q[0] = *(const uint4*)ap;
            fa[mi].q[1] = *(const uint4*)(ap + 16);
        }
        // B fragments straight from L2-resident packed W2
        #pragma unroll
        for (int ni = 0; ni < 2; ++ni) {
            const __bf16* bp = w2pack + ((((size_t)(ut0 + ni)) * 32 + it) * 32 + lane) * 16;
            fb[ni].q[0] = *(const uint4*)bp;
            fb[ni].q[1] = *(const uint4*)(bp + 8);
        }

        #pragma unroll
        for (int mi = 0; mi < 2; ++mi)
            #pragma unroll
            for (int ni = 0; ni < 2; ++ni)
                c[mi][ni] = __builtin_amdgcn_wmma_f32_16x16x32_bf16(
                    false, fa[mi].v, false, fb[ni].v, (short)0, c[mi][ni], false, false);
    }

    // epilogue: tanh(qproj + vproj + b2) * V_k, reduce this wave's 32 N columns
    float qadd[2], vkv[2];
    #pragma unroll
    for (int ni = 0; ni < 2; ++ni) {
        int ng = ucol0 + nw + ni * 16 + lrow;
        qadd[ni] = qproj[(size_t)b * U_ + ng] + w2b[ng];
        vkv[ni]  = vk[ng];
    }
    #pragma unroll
    for (int mi = 0; mi < 2; ++mi) {
        #pragma unroll
        for (int r = 0; r < 8; ++r) {
            float pr = tanhf(c[mi][0][r] + qadd[0]) * vkv[0]
                     + tanhf(c[mi][1][r] + qadd[1]) * vkv[1];
            pr += __shfl_xor(pr, 1, 32);
            pr += __shfl_xor(pr, 2, 32);
            pr += __shfl_xor(pr, 4, 32);
            pr += __shfl_xor(pr, 8, 32);   // sum over 16-lane half-groups
            if (lrow == 0) {
                int m = mw + mi * 16 + r + ((lane >> 4) << 3);
                unsafeAtomicAdd(&score[(size_t)b * T_ + trow0 + m], pr);
            }
        }
    }
}

// ---------------------------------------------------------------- softmax(T)
__global__ __launch_bounds__(256) void softmax_kernel(float* __restrict__ score,
                                                      const float* __restrict__ Vb) {
    const int b = blockIdx.x, tid = threadIdx.x;
    __shared__ float red[256];
    float s[8];
    const float vb = Vb[0];
    float mx = -3.0e38f;
    #pragma unroll
    for (int i = 0; i < 8; ++i) {
        s[i] = score[(size_t)b * T_ + tid + i * 256] + vb;
        mx = fmaxf(mx, s[i]);
    }
    red[tid] = mx; __syncthreads();
    for (int off = 128; off; off >>= 1) {
        if (tid < off) red[tid] = fmaxf(red[tid], red[tid + off]);
        __syncthreads();
    }
    mx = red[0]; __syncthreads();

    float sum = 0.0f;
    #pragma unroll
    for (int i = 0; i < 8; ++i) { s[i] = __expf(s[i] - mx); sum += s[i]; }
    red[tid] = sum; __syncthreads();
    for (int off = 128; off; off >>= 1) {
        if (tid < off) red[tid] += red[tid + off];
        __syncthreads();
    }
    const float inv = 1.0f / red[0];
    #pragma unroll
    for (int i = 0; i < 8; ++i)
        score[(size_t)b * T_ + tid + i * 256] = s[i] * inv;
}

// -------------------------------------------------- context = sum_t w_t * v_t
__global__ __launch_bounds__(256) void context_kernel(const float* __restrict__ values,
                                                      const float* __restrict__ attn,
                                                      float* __restrict__ out) {
    const int b  = blockIdx.x;
    const int t0 = blockIdx.y * 256;          // 8 chunks of 256 timesteps
    const int d0 = threadIdx.x * 4;
    float4 acc = make_float4(0.f, 0.f, 0.f, 0.f);
    for (int t = t0; t < t0 + 256; ++t) {
        const float w = attn[(size_t)b * T_ + t];
        const float4 v = *(const float4*)(values + ((size_t)b * T_ + t) * D_ + d0);
        acc.x = fmaf(w, v.x, acc.x);
        acc.y = fmaf(w, v.y, acc.y);
        acc.z = fmaf(w, v.z, acc.z);
        acc.w = fmaf(w, v.w, acc.w);
    }
    float* o = out + (size_t)b * D_ + d0;
    unsafeAtomicAdd(o + 0, acc.x);
    unsafeAtomicAdd(o + 1, acc.y);
    unsafeAtomicAdd(o + 2, acc.z);
    unsafeAtomicAdd(o + 3, acc.w);
}

// ------------------------------------------------------------------- launch
extern "C" void kernel_launch(void* const* d_in, const int* in_sizes, int n_in,
                              void* d_out, int out_size, void* d_ws, size_t ws_size,
                              hipStream_t stream) {
    const float* query = (const float*)d_in[0];
    const float* values = (const float*)d_in[1];
    const float* W1     = (const float*)d_in[2];
    const float* W1b    = (const float*)d_in[3];
    const float* W2     = (const float*)d_in[4];
    const float* W2b    = (const float*)d_in[5];
    const float* Vk     = (const float*)d_in[6];
    const float* Vb     = (const float*)d_in[7];
    float* out  = (float*)d_out;

    float*  ws_f   = (float*)d_ws;
    float*  score  = ws_f;                        // B*T  floats (256 KB)
    float*  qproj  = ws_f + (size_t)B_ * T_;      // B*U  floats (128 KB)
    __bf16* w2pack = (__bf16*)(ws_f + (size_t)B_ * T_ + (size_t)B_ * U_);  // D*U bf16 (2 MB)

    zero_kernel   <<<96,  256, 0, stream>>>(score, out);
    pack_w2_kernel<<<256, 256, 0, stream>>>(W2, w2pack);
    qproj_kernel  <<<128, 256, 0, stream>>>(query, W1, W1b, qproj);

    dim3 g(U_ / 128, T_ / 64, B_);
    bahdanau_score_wmma<<<g, 256, 0, stream>>>(values, w2pack, qproj, W2b, Vk, score);

    softmax_kernel<<<B_, 256, 0, stream>>>(score, Vb);

    dim3 gc(B_, 8);
    context_kernel<<<gc, 256, 0, stream>>>(values, score, out);
}
